// PhiMoESparseMoeBlock_12266426597722
// MI455X (gfx1250) — matrix operand order repaired and checked by reading, not compiled, not run on hardware
//
#include <hip/hip_runtime.h>
#include <hip/hip_bf16.h>
#include <cstdint>

#define TOK   1024     // B*S
#define HDIM  2048
#define IDIM  4096
#define NEXP  8

typedef __attribute__((ext_vector_type(16))) __bf16 v16bf;
typedef __attribute__((ext_vector_type(8)))  __bf16 v8bf;
typedef __attribute__((ext_vector_type(8)))  float  v8f;

// Pack two fp32 -> packed bf16 pair (native converts; backend can fuse to
// v_cvt_pk_bf16_f32). RNE rounding matches jnp/bf16 semantics closely enough.
__device__ __forceinline__ unsigned cvt2bf(float x, float y) {
    union { __bf16 b[2]; unsigned u; } r;
    r.b[0] = (__bf16)x;
    r.b[1] = (__bf16)y;
    return r.u;
}

// A-matrix tile load (16x32 bf16): lane<16 holds K[0..15];
// lane>=16 holds K[16..31]. Pointer pre-offset by (lane>=16 ? 16 : 0).
// Union-combine: the two 16B loads land in adjacent VGPRs, no half-moves.
__device__ __forceinline__ v16bf load_a16(const __bf16* p) {
    union { v8bf h[2]; v16bf v; } r;
    r.h[0] = *(const v8bf*)(p);
    r.h[1] = *(const v8bf*)(p + 8);
    return r.v;
}

// B-matrix tile load (32x16 bf16): lane<16 -> K[0..15], lane>=16 -> K[16..31],
// N = lane&15. Pointer pre-offset by row*ld + (lane>=16 ? 16 : 0). fp32 -> bf16.
__device__ __forceinline__ v16bf load_b16(const float* p) {
    union { unsigned u[8]; v16bf v; } r;
#pragma unroll
    for (int i = 0; i < 4; ++i) {
        float4 f = ((const float4*)p)[i];
        r.u[2 * i + 0] = cvt2bf(f.x, f.y);
        r.u[2 * i + 1] = cvt2bf(f.z, f.w);
    }
    return r.v;
}

__device__ __forceinline__ v8f wmma_bf16(v16bf a, v16bf b, v8f c) {
    return __builtin_amdgcn_wmma_f32_16x16x32_bf16(false, a, false, b, (short)0, c,
                                                   false, false);
}

__device__ __forceinline__ float fast_silu(float z) {
    // z * sigmoid(z) with single v_rcp_f32
    return z * __builtin_amdgcn_rcpf(1.f + __expf(-z));
}

// ---------------- kernel 1: fp32 X -> bf16 X ----------------
__global__ __launch_bounds__(256) void k_cvt_x(const float* __restrict__ x,
                                               __bf16* __restrict__ xb) {
    int i = (blockIdx.x * 256 + threadIdx.x) * 8;
    float4 f0 = *(const float4*)(x + i);
    float4 f1 = *(const float4*)(x + i + 4);
    union { unsigned u[4]; v8bf v; } o;
    o.u[0] = cvt2bf(f0.x, f0.y);
    o.u[1] = cvt2bf(f0.z, f0.w);
    o.u[2] = cvt2bf(f1.x, f1.y);
    o.u[3] = cvt2bf(f1.z, f1.w);
    *(v8bf*)(xb + i) = o.v;
}

// ---------------- kernel 2: gate logits + sparsemixer top-2 routing ----------------
__global__ __launch_bounds__(256) void k_gate(const float* __restrict__ x,
                                              const float* __restrict__ gw,
                                              int2* __restrict__ sel,
                                              float2* __restrict__ mult) {
    __shared__ float s_lg[NEXP];
    const int t    = blockIdx.x;
    const int lane = threadIdx.x & 31;
    const int e    = threadIdx.x >> 5;          // 8 waves -> 8 experts
    const float* xr = x  + (size_t)t * HDIM;
    const float* gr = gw + (size_t)e * HDIM;
    float acc = 0.f;
    for (int h = lane; h < HDIM; h += 32) acc += xr[h] * gr[h];
#pragma unroll
    for (int off = 16; off > 0; off >>= 1) acc += __shfl_down(acc, off, 32);
    if (lane == 0) s_lg[e] = acc;
    __syncthreads();
    if (threadIdx.x == 0) {
        float lg[NEXP];
#pragma unroll
        for (int i = 0; i < NEXP; ++i) lg[i] = s_lg[i];
        const float TH2 = 2.f * 0.01f;          // 2 * JITTER_EPS
        // top-1
        float max1 = -INFINITY; int i1 = 0;
#pragma unroll
        for (int i = 0; i < NEXP; ++i) if (lg[i] > max1) { max1 = lg[i]; i1 = i; }
        float s1 = 0.f;
#pragma unroll
        for (int i = 0; i < NEXP; ++i) {
            float fac = fmaxf(fabsf(lg[i]), max1);
            if (!(((max1 - lg[i]) / fac) > TH2)) s1 += __expf(lg[i] - max1);
        }
        float m1 = 1.f / s1;                    // gates1[i1]
        // top-2 (mask out i1)
        float max2 = -INFINITY; int i2 = 0;
#pragma unroll
        for (int i = 0; i < NEXP; ++i)
            if (i != i1 && lg[i] > max2) { max2 = lg[i]; i2 = i; }
        float s2 = 0.f;
#pragma unroll
        for (int i = 0; i < NEXP; ++i) {
            if (i == i1) continue;
            float fac = fmaxf(fabsf(lg[i]), max2);
            if (!(((max2 - lg[i]) / fac) > TH2)) s2 += __expf(lg[i] - max2);
        }
        float m2 = 1.f / s2;                    // gates2[i2]
        sel[t]  = make_int2(i1, i2);
        mult[t] = make_float2(m1, m2);
    }
}

// ---------------- kernel 3: compact (token,k) pairs per expert ----------------
__global__ __launch_bounds__(256) void k_scatter(const int2* __restrict__ sel,
                                                 const float2* __restrict__ mult,
                                                 int* __restrict__ cnt,
                                                 int* __restrict__ plist,
                                                 float* __restrict__ wlist) {
    int t = blockIdx.x * 256 + threadIdx.x;
    if (t >= TOK) return;
    int2 s = sel[t]; float2 m = mult[t];
    int pos = atomicAdd(&cnt[s.x], 1);
    plist[s.x * TOK + pos] = 2 * t;     wlist[s.x * TOK + pos] = m.x;
    pos = atomicAdd(&cnt[s.y], 1);
    plist[s.y * TOK + pos] = 2 * t + 1; wlist[s.y * TOK + pos] = m.y;
}

// ---------------- kernel 4: act = silu(X@w1^T) * (X@w3^T) per expert ----------------
// Block: 256 thr (8 waves). Wave -> 32 tokens x 32 I outputs (2x2 WMMA tiles, x2 mats).
// grid = (Mtiles=32, IDIM/256=16, NEXP); blockIdx.x fastest for L2 weight reuse.
__global__ __launch_bounds__(256) void k_mlp1(const __bf16* __restrict__ xb,
                                              const float* __restrict__ w1,
                                              const float* __restrict__ w3,
                                              const int* __restrict__ plist,
                                              const int* __restrict__ cnt,
                                              __bf16* __restrict__ act) {
    const int e  = blockIdx.z;
    const int ne = cnt[e];
    const int m0 = blockIdx.x * 32;
    if (m0 >= ne) return;
    __shared__ int s_p[32];
    if (threadIdx.x < 32) {
        int r = m0 + threadIdx.x;
        s_p[threadIdx.x] = (r < ne) ? plist[e * TOK + r] : -1;
    }
    __syncthreads();
    const int lane = threadIdx.x & 31;
    const int wv   = threadIdx.x >> 5;
    const int hl   = lane >> 4;      // lane half
    const int l15  = lane & 15;
    // A row pointers (gathered tokens), pre-offset by hl*16 (contiguous k = 16*hl + s)
    int pA0 = s_p[l15], pA1 = s_p[16 + l15];
    const __bf16* a0p = xb + (size_t)(pA0 < 0 ? 0 : (pA0 >> 1)) * HDIM + hl * 16;
    const __bf16* a1p = xb + (size_t)(pA1 < 0 ? 0 : (pA1 >> 1)) * HDIM + hl * 16;
    // B row pointers: w[e, i, :] row-major over H
    const int i0 = blockIdx.y * 256 + wv * 32;
    const int bk = hl * 16;
    const size_t wbase = (size_t)e * IDIM * HDIM;
    const float* b1p0 = w1 + wbase + (size_t)(i0 + l15) * HDIM + bk;
    const float* b1p1 = w1 + wbase + (size_t)(i0 + 16 + l15) * HDIM + bk;
    const float* b3p0 = w3 + wbase + (size_t)(i0 + l15) * HDIM + bk;
    const float* b3p1 = w3 + wbase + (size_t)(i0 + 16 + l15) * HDIM + bk;

    const v8f zero = {0.f, 0.f, 0.f, 0.f, 0.f, 0.f, 0.f, 0.f};
    v8f c1[2][2], c3[2][2];
#pragma unroll
    for (int m = 0; m < 2; ++m)
#pragma unroll
        for (int n = 0; n < 2; ++n) { c1[m][n] = zero; c3[m][n] = zero; }

    for (int k = 0; k < HDIM; k += 32) {
        __builtin_prefetch(b1p0 + k + 256, 0, 1);
        __builtin_prefetch(b3p0 + k + 256, 0, 1);
        v16bf a0  = load_a16(a0p + k);
        v16bf a1  = load_a16(a1p + k);
        v16bf b10 = load_b16(b1p0 + k);
        v16bf b11 = load_b16(b1p1 + k);
        v16bf b30 = load_b16(b3p0 + k);
        v16bf b31 = load_b16(b3p1 + k);
        c1[0][0] = wmma_bf16(a0, b10, c1[0][0]);
        c1[0][1] = wmma_bf16(a0, b11, c1[0][1]);
        c1[1][0] = wmma_bf16(a1, b10, c1[1][0]);
        c1[1][1] = wmma_bf16(a1, b11, c1[1][1]);
        c3[0][0] = wmma_bf16(a0, b30, c3[0][0]);
        c3[0][1] = wmma_bf16(a0, b31, c3[0][1]);
        c3[1][0] = wmma_bf16(a1, b30, c3[1][0]);
        c3[1][1] = wmma_bf16(a1, b31, c3[1][1]);
    }
    // epilogue: silu(h1) * h3 -> act (bf16). C layout: M = j + 8*hl, N = l15.
    // Row guard hoisted out of the n-loop (row independent of n).
#pragma unroll
    for (int m = 0; m < 2; ++m)
#pragma unroll
        for (int j = 0; j < 8; ++j) {
            int row = m * 16 + hl * 8 + j;
            int p = s_p[row];
            if (p < 0) continue;
            __bf16* arow = act + (size_t)p * IDIM + i0 + l15;
            float v0 = fast_silu(c1[m][0][j]) * c3[m][0][j];
            float v1 = fast_silu(c1[m][1][j]) * c3[m][1][j];
            arow[0]  = (__bf16)v0;
            arow[16] = (__bf16)v1;
        }
}

// ---------------- kernel 5: out += wt * (act @ w2^T) ----------------
// grid = (Mtiles=32, HDIM/256=8, NEXP)
__global__ __launch_bounds__(256) void k_mlp2(const __bf16* __restrict__ act,
                                              const float* __restrict__ w2,
                                              const int* __restrict__ plist,
                                              const float* __restrict__ wlist,
                                              const int* __restrict__ cnt,
                                              float* __restrict__ out) {
    const int e  = blockIdx.z;
    const int ne = cnt[e];
    const int m0 = blockIdx.x * 32;
    if (m0 >= ne) return;
    __shared__ int   s_p[32];
    __shared__ float s_w[32];
    if (threadIdx.x < 32) {
        int r = m0 + threadIdx.x;
        bool v = (r < ne);
        s_p[threadIdx.x] = v ? plist[e * TOK + r] : -1;
        s_w[threadIdx.x] = v ? wlist[e * TOK + r] : 0.f;
    }
    __syncthreads();
    const int lane = threadIdx.x & 31;
    const int wv   = threadIdx.x >> 5;
    const int hl   = lane >> 4;
    const int l15  = lane & 15;
    int pA0 = s_p[l15], pA1 = s_p[16 + l15];
    const __bf16* a0p = act + (size_t)(pA0 < 0 ? 0 : pA0) * IDIM + hl * 16;
    const __bf16* a1p = act + (size_t)(pA1 < 0 ? 0 : pA1) * IDIM + hl * 16;
    // B: w2[e, h, :] row-major over I
    const int h0 = blockIdx.y * 256 + wv * 32;
    const int bk = hl * 16;
    const size_t wbase = (size_t)e * HDIM * IDIM;
    const float* b0p = w2 + wbase + (size_t)(h0 + l15) * IDIM + bk;
    const float* b1p = w2 + wbase + (size_t)(h0 + 16 + l15) * IDIM + bk;

    const v8f zero = {0.f, 0.f, 0.f, 0.f, 0.f, 0.f, 0.f, 0.f};
    v8f c[2][2];
#pragma unroll
    for (int m = 0; m < 2; ++m)
#pragma unroll
        for (int n = 0; n < 2; ++n) c[m][n] = zero;

    for (int k = 0; k < IDIM; k += 32) {
        __builtin_prefetch(b0p + k + 256, 0, 1);
        v16bf a0 = load_a16(a0p + k);
        v16bf a1 = load_a16(a1p + k);
        v16bf b0 = load_b16(b0p + k);
        v16bf b1 = load_b16(b1p + k);
        c[0][0] = wmma_bf16(a0, b0, c[0][0]);
        c[0][1] = wmma_bf16(a0, b1, c[0][1]);
        c[1][0] = wmma_bf16(a1, b0, c[1][0]);
        c[1][1] = wmma_bf16(a1, b1, c[1][1]);
    }
    // scatter: out[t, h] += wt * y   (exactly 2 commutative adds per element)
#pragma unroll
    for (int m = 0; m < 2; ++m)
#pragma unroll
        for (int j = 0; j < 8; ++j) {
            int row = m * 16 + hl * 8 + j;
            int p = s_p[row];
            if (p < 0) continue;
            float wt = s_w[row];
            float* orow = out + (size_t)(p >> 1) * HDIM + h0 + l15;
            atomicAdd(orow,      c[m][0][j] * wt);
            atomicAdd(orow + 16, c[m][1][j] * wt);
        }
}

extern "C" void kernel_launch(void* const* d_in, const int* in_sizes, int n_in,
                              void* d_out, int out_size, void* d_ws, size_t ws_size,
                              hipStream_t stream) {
    const float* x  = (const float*)d_in[0];   // [T, H]
    const float* gw = (const float*)d_in[1];   // [E, H]
    const float* w1 = (const float*)d_in[2];   // [E, I, H]
    const float* w2 = (const float*)d_in[3];   // [E, H, I]
    const float* w3 = (const float*)d_in[4];   // [E, I, H]
    float* out = (float*)d_out;

    // workspace layout (~21 MB)
    char* ws = (char*)d_ws;
    __bf16* xb  = (__bf16*)(ws);                              // 4 MB
    __bf16* act = (__bf16*)(ws + (size_t)4 * 1024 * 1024);    // 16 MB
    size_t o = (size_t)20 * 1024 * 1024;
    int*    plist = (int*)   (ws + o); o += (size_t)NEXP * TOK * 4;
    float*  wlist = (float*) (ws + o); o += (size_t)NEXP * TOK * 4;
    int2*   sel   = (int2*)  (ws + o); o += (size_t)TOK * 8;
    float2* mult  = (float2*)(ws + o); o += (size_t)TOK * 8;
    int*    cnt   = (int*)   (ws + o); o += 256;

    hipMemsetAsync(cnt, 0, 256, stream);
    hipMemsetAsync(out, 0, (size_t)TOK * HDIM * sizeof(float), stream);

    k_cvt_x  <<<(TOK * HDIM) / (8 * 256), 256, 0, stream>>>(x, xb);
    k_gate   <<<TOK, 256, 0, stream>>>(x, gw, sel, mult);
    k_scatter<<<(TOK + 255) / 256, 256, 0, stream>>>(sel, mult, cnt, plist, wlist);
    k_mlp1   <<<dim3(TOK / 32, IDIM / 256, NEXP), 256, 0, stream>>>(xb, w1, w3,
                                                                    plist, cnt, act);
    k_mlp2   <<<dim3(TOK / 32, HDIM / 256, NEXP), 256, 0, stream>>>(act, w2, plist,
                                                                    wlist, cnt, out);
}
